// GraphEmbedding_33938831573347
// MI455X (gfx1250) — matrix operand order repaired and checked by reading, not compiled, not run on hardware
//
#include <hip/hip_runtime.h>
#include <cstdint>
#include <cstddef>

typedef _Float16 half_t;
typedef __attribute__((ext_vector_type(16))) _Float16 v16h;
typedef __attribute__((ext_vector_type(8)))  _Float16 v8h;
typedef __attribute__((ext_vector_type(4)))  _Float16 v4h;
typedef __attribute__((ext_vector_type(8)))  float    v8f;
typedef __attribute__((ext_vector_type(4)))  float    v4f;

#define KNBR  20
#define DFEAT 128
#define FIN   384            // D + DT + DE
#define LDSTR (FIN + 8)      // 392 halfs per row -> 784B stride, 16B aligned
#define WPER  (FIN * DFEAT)  // 49152 elements per weight matrix

// ---------------------------------------------------------------------------
// prep: swizzle W1/W2 (both layers) f32->f16 into WMMA B-fragment-major
// layout, and compute stb = cos(time_b).
// Packed index: P[((kc*8 + colTile)*32 + lane)*16 + j]
//             = W[(kc*32 + (lane>>4)*16 + j)*128 + colTile*16 + (lane&15)]
// so each lane's 16-half fragment is one contiguous 32-byte load.
// ---------------------------------------------------------------------------
__global__ void tgn_prep_kernel(const float* __restrict__ W1,
                                const float* __restrict__ W2,
                                const float* __restrict__ time_b,
                                half_t* __restrict__ w1h,
                                half_t* __restrict__ w2h,
                                half_t* __restrict__ stbh)
{
    int idx = blockIdx.x * blockDim.x + threadIdx.x;
    if (idx < 4 * WPER) {
        int mat = idx / WPER;           // 0,1: W1 l0/l1 ; 2,3: W2 l0/l1
        int e   = idx - mat * WPER;
        int j   = e & 15;
        int L   = (e >> 4) & 31;
        int ct  = (e >> 9) & 7;
        int kc  = e >> 12;              // 0..11
        int srcRow = kc * 32 + (L >> 4) * 16 + j;
        int srcCol = ct * 16 + (L & 15);
        const float* src = (mat < 2) ? (W1 + (size_t)mat * WPER)
                                     : (W2 + (size_t)(mat - 2) * WPER);
        half_t* dst = (mat < 2) ? (w1h + (size_t)mat * WPER)
                                : (w2h + (size_t)(mat - 2) * WPER);
        dst[e] = (half_t)src[(size_t)srcRow * DFEAT + srcCol];
    } else if (idx < 4 * WPER + DFEAT) {
        int c = idx - 4 * WPER;
        stbh[c] = (half_t)cosf(time_b[c]);
    }
}

// ---------------------------------------------------------------------------
// gather: SNF[b, 0:384] = sum_k mask * [scale*nbr_tab | cos(dt*w+b) | ef]
// one wave32 per row; each lane owns 4 consecutive columns of each 128-wide
// segment -> full-width b128 transactions on the gathered rows.
// ---------------------------------------------------------------------------
__global__ void tgn_snf_gather_kernel(
    const float* __restrict__ nbr_table,      // memory (gathered) or out2 (sequential)
    const float* __restrict__ edge_features,
    const float* __restrict__ timestamps,
    const float* __restrict__ time_w,
    const float* __restrict__ time_b,
    const int*   __restrict__ nbr,            // [B,K] ids (also the mask source)
    const int*   __restrict__ eidx,           // [B,K]
    const float* __restrict__ etime,          // [B,K]
    half_t* __restrict__ snf,                 // [B,384] f16 out
    int B, int ts_div, float nbr_scale, int nbr_seq)
{
    int wave = blockIdx.x * (blockDim.x >> 5) + (threadIdx.x >> 5);
    int lane = threadIdx.x & 31;
    if (wave >= B) return;
    const int b = wave;
    const float ts = timestamps[b / ts_div];

    const v4f tw = ((const v4f*)time_w)[lane];
    const v4f tb = ((const v4f*)time_b)[lane];
    v4f accN = {}, accT = {}, accE = {};

    for (int k = 0; k < KNBR; ++k) {
        int nb = nbr[(size_t)b * KNBR + k];
        if (nb == 0) continue;                       // mask: all 384 feats zeroed
        int   ei = eidx[(size_t)b * KNBR + k];
        float et = etime[(size_t)b * KNBR + k];
        float dt = ts - et;
        const float* nrow = nbr_table +
            (size_t)(nbr_seq ? (b * KNBR + k) : nb) * DFEAT;
        const float* erow = edge_features + (size_t)ei * DFEAT;
        v4f nv = ((const v4f*)nrow)[lane];
        v4f ev = ((const v4f*)erow)[lane];
#pragma unroll
        for (int i = 0; i < 4; ++i) {
            accN[i] += nbr_scale * nv[i];
            accT[i] += cosf(dt * tw[i] + tb[i]);
            accE[i] += ev[i];
        }
    }

    half_t* orow = snf + (size_t)b * FIN;
    v4h hN, hT, hE;
#pragma unroll
    for (int i = 0; i < 4; ++i) {
        hN[i] = (half_t)accN[i];
        hT[i] = (half_t)accT[i];
        hE[i] = (half_t)accE[i];
    }
    ((v4h*)orow)[lane]                 = hN;
    ((v4h*)(orow + DFEAT))[lane]       = hT;
    ((v4h*)(orow + 2 * DFEAT))[lane]   = hE;
}

// ---------------------------------------------------------------------------
// fused layer: s = relu(SNF @ W1 + K*b1); out = [s | src | stb] @ W2 + b2
// 256 threads = 8 wave32; block handles 16 rows x 128 cols; each wave owns a
// 16-col slice and iterates 12 K-chunks of the 16x16x32 f16 WMMA.
// Weights arrive pre-swizzled: one v16h (32B) load per lane per chunk.
// ---------------------------------------------------------------------------
__global__ void tgn_layer_gemm_kernel(
    const half_t* __restrict__ snf,        // [B,384] f16
    const half_t* __restrict__ w1,         // [12*8*32*16] swizzled f16
    const float*  __restrict__ b1,         // [128]
    const half_t* __restrict__ w2,         // swizzled f16
    const float*  __restrict__ b2,         // [128]
    const float*  __restrict__ src_table,  // memory / conv
    const int*    __restrict__ src_idx,    // null -> sequential rows
    const half_t* __restrict__ stb,        // [128] f16 cos(time_b)
    float* __restrict__ out,               // [B,128]
    int B, float src_scale, float kcnt)
{
    __shared__ __align__(16) half_t lA[16 * LDSTR];   // SNF tile
    __shared__ __align__(16) half_t lC[16 * LDSTR];   // src_cat tile

    const int t    = threadIdx.x;
    const int wave = t >> 5;
    const int lane = t & 31;
    const int row0 = blockIdx.x * 16;

    // stage SNF tile (f16) into LDS: 16B granular copies
    for (int idx = t; idx < 16 * (FIN / 8); idx += 256) {
        int r = idx / (FIN / 8), c = (idx - r * (FIN / 8)) * 8;
        *(v8h*)&lA[r * LDSTR + c] = *(const v8h*)&snf[(size_t)(row0 + r) * FIN + c];
    }
    // src_cat cols 128..255: src embedding (float4 -> 4 halfs)
    for (int idx = t; idx < 16 * (DFEAT / 4); idx += 256) {
        int r = idx / (DFEAT / 4), c = (idx - r * (DFEAT / 4)) * 4;
        int row = row0 + r;
        const float* srow = src_idx ? (src_table + (size_t)src_idx[row] * DFEAT)
                                    : (src_table + (size_t)row * DFEAT);
        v4f sv = *(const v4f*)&srow[c];
        v4h hv;
#pragma unroll
        for (int i = 0; i < 4; ++i) hv[i] = (half_t)(src_scale * sv[i]);
        *(v4h*)&lC[r * LDSTR + DFEAT + c] = hv;
    }
    // src_cat cols 256..383: constant time encoding
    for (int idx = t; idx < 16 * (DFEAT / 8); idx += 256) {
        int r = idx / (DFEAT / 8), c = (idx - r * (DFEAT / 8)) * 8;
        *(v8h*)&lC[r * LDSTR + 2 * DFEAT + c] = *(const v8h*)&stb[c];
    }
    __syncthreads();

    const int m    = lane & 15;
    const int hi16 = lane >> 4;        // 0 | 1
    const int koff = hi16 * 16;        // A-fragment K base within chunk
    const int n    = lane & 15;
    const int col0 = wave * 16;

    // ---- GEMM1: SNF @ W1 ----
    v8f acc = {};
    for (int kc = 0; kc < FIN / 32; ++kc) {
        int base = kc * 32 + koff;
        v8h alo = *(const v8h*)&lA[m * LDSTR + base];
        v8h ahi = *(const v8h*)&lA[m * LDSTR + base + 8];
        v16h a;
#pragma unroll
        for (int i = 0; i < 8; ++i) { a[i] = alo[i]; a[8 + i] = ahi[i]; }
        v16h bf = *(const v16h*)&w1[(size_t)(((kc * 8 + wave) * 32 + lane)) * 16];
        acc = __builtin_amdgcn_wmma_f32_16x16x32_f16(
                  false, a, false, bf, (short)0, acc, false, false);
    }
    {   // epilogue: + K*b1, ReLU, write s slice into src_cat cols [col0,col0+16)
        float bias = kcnt * b1[col0 + n];
#pragma unroll
        for (int r = 0; r < 8; ++r) {
            int mm = r + 8 * hi16;
            float s = acc[r] + bias;
            s = s > 0.0f ? s : 0.0f;
            lC[mm * LDSTR + col0 + n] = (half_t)s;
        }
    }
    __syncthreads();

    // ---- GEMM2: src_cat @ W2 ----
    v8f acc2 = {};
    for (int kc = 0; kc < FIN / 32; ++kc) {
        int base = kc * 32 + koff;
        v8h alo = *(const v8h*)&lC[m * LDSTR + base];
        v8h ahi = *(const v8h*)&lC[m * LDSTR + base + 8];
        v16h a;
#pragma unroll
        for (int i = 0; i < 8; ++i) { a[i] = alo[i]; a[8 + i] = ahi[i]; }
        v16h bf = *(const v16h*)&w2[(size_t)(((kc * 8 + wave) * 32 + lane)) * 16];
        acc2 = __builtin_amdgcn_wmma_f32_16x16x32_f16(
                   false, a, false, bf, (short)0, acc2, false, false);
    }
    {
        float bias = b2[col0 + n];
#pragma unroll
        for (int r = 0; r < 8; ++r) {
            int mm = r + 8 * hi16;
            out[(size_t)(row0 + mm) * DFEAT + col0 + n] = acc2[r] + bias;
        }
    }
}

// ---------------------------------------------------------------------------
extern "C" void kernel_launch(void* const* d_in, const int* in_sizes, int n_in,
                              void* d_out, int out_size, void* d_ws, size_t ws_size,
                              hipStream_t stream)
{
    (void)in_sizes; (void)n_in; (void)out_size; (void)ws_size;

    const float* memory        = (const float*)d_in[0];
    const float* edge_features = (const float*)d_in[1];
    const float* timestamps    = (const float*)d_in[2];
    const float* time_w        = (const float*)d_in[3];
    const float* time_b        = (const float*)d_in[4];
    const float* W1            = (const float*)d_in[5];
    const float* b1            = (const float*)d_in[6];
    const float* W2            = (const float*)d_in[7];
    const float* b2            = (const float*)d_in[8];
    const int*   source_nodes  = (const int*)d_in[9];
    const int*   nbr1          = (const int*)d_in[10];
    const int*   eidx1         = (const int*)d_in[11];
    const float* etime1        = (const float*)d_in[12];
    const int*   nbr2          = (const int*)d_in[13];
    const int*   eidx2         = (const int*)d_in[14];
    const float* etime2        = (const float*)d_in[15];

    const int N  = 1024;
    const int K  = KNBR;
    const int B2 = N * K;     // 20480

    char* ws = (char*)d_ws;
    size_t off = 0;
    auto take = [&](size_t bytes) -> char* {
        char* p = ws + off;
        off = (off + bytes + 255) & ~(size_t)255;
        return p;
    };
    half_t* w1h  = (half_t*)take((size_t)2 * WPER * sizeof(half_t));
    half_t* w2h  = (half_t*)take((size_t)2 * WPER * sizeof(half_t));
    half_t* stbh = (half_t*)take((size_t)DFEAT * sizeof(half_t));
    half_t* snf2 = (half_t*)take((size_t)B2 * FIN * sizeof(half_t));
    half_t* snf1 = (half_t*)take((size_t)N  * FIN * sizeof(half_t));
    half_t* snfc = (half_t*)take((size_t)N  * FIN * sizeof(half_t));
    float*  out2 = (float*) take((size_t)B2 * DFEAT * sizeof(float));
    float*  conv = (float*) take((size_t)N  * DFEAT * sizeof(float));

    // 0) weight swizzle + cos(time_b)
    {
        int total = 4 * WPER + DFEAT;
        tgn_prep_kernel<<<(total + 255) / 256, 256, 0, stream>>>(
            W1, W2, time_b, w1h, w2h, stbh);
    }

    // --- Stage A: layer-1 on all 1-hop neighbors (B2 rows, layer 0) ---
    tgn_snf_gather_kernel<<<(B2 + 7) / 8, 256, 0, stream>>>(
        memory, edge_features, timestamps, time_w, time_b,
        nbr2, eidx2, etime2, snf2, B2, /*ts_div=*/K, /*scale=*/2.0f, /*seq=*/0);
    tgn_layer_gemm_kernel<<<B2 / 16, 256, 0, stream>>>(
        snf2, w1h, b1, w2h, b2,
        memory, nbr1, stbh, out2, B2, /*src_scale=*/2.0f, (float)K);

    // --- Stage B: layer-1 on sources (N rows, layer 0) -> conv ---
    tgn_snf_gather_kernel<<<(N + 7) / 8, 256, 0, stream>>>(
        memory, edge_features, timestamps, time_w, time_b,
        nbr1, eidx1, etime1, snf1, N, /*ts_div=*/1, /*scale=*/2.0f, /*seq=*/0);
    tgn_layer_gemm_kernel<<<N / 16, 256, 0, stream>>>(
        snf1, w1h, b1, w2h, b2,
        memory, source_nodes, stbh, conv, N, /*src_scale=*/2.0f, (float)K);

    // --- Stage C: layer-2 aggregation (N rows, layer 1) -> output ---
    tgn_snf_gather_kernel<<<(N + 7) / 8, 256, 0, stream>>>(
        out2 /*nbr_emb_l1, sequential*/, edge_features, timestamps, time_w, time_b,
        nbr1, eidx1, etime1, snfc, N, /*ts_div=*/1, /*scale=*/1.0f, /*seq=*/1);
    tgn_layer_gemm_kernel<<<N / 16, 256, 0, stream>>>(
        snfc, w1h + (size_t)WPER, b1 + DFEAT,
        w2h + (size_t)WPER, b2 + DFEAT,
        conv, /*src_idx=*/nullptr, stbh, (float*)d_out, N,
        /*src_scale=*/1.0f, (float)K);
}
